// RNN_7988639170692
// MI455X (gfx1250) — matrix operand order repaired and checked by reading, not compiled, not run on hardware
//
#include <hip/hip_runtime.h>

// ---------------------------------------------------------------------------
// Problem constants (from the reference)
// ---------------------------------------------------------------------------
constexpr int S_ = 64;     // sequence length
constexpr int B_ = 32;     // batch
constexpr int H_ = 1024;   // hidden == emb
constexpr int V_ = 32000;  // vocab

typedef __bf16 bf16;
typedef __attribute__((ext_vector_type(16))) __bf16 v16bf;
typedef __attribute__((ext_vector_type(8)))  __bf16 v8bf;
typedef __attribute__((ext_vector_type(8)))  float  v8f;
typedef __attribute__((ext_vector_type(4)))  unsigned int u32x4;
typedef __attribute__((ext_vector_type(8)))  int i32x8;
typedef __attribute__((ext_vector_type(4)))  int i32x4;
typedef __attribute__((address_space(3)))    bf16 bf16_lds;

union ABf { struct { v8bf lo, hi; } h; v16bf v; };

__device__ __forceinline__ bf16 f2bf(float f) {
  unsigned u = __builtin_bit_cast(unsigned, f);
  u += 0x7fffu + ((u >> 16) & 1u);             // round-to-nearest-even
  unsigned short s = (unsigned short)(u >> 16);
  return __builtin_bit_cast(bf16, s);
}

__device__ __forceinline__ v8f vzero() {
  v8f z = {0.f, 0.f, 0.f, 0.f, 0.f, 0.f, 0.f, 0.f};
  return z;
}

__device__ __forceinline__ v8f wmma_bf(v16bf a, v16bf b, v8f c) {
  // D = A(16x32 bf16) x B(32x16 bf16) + C(16x16 f32)
  return __builtin_amdgcn_wmma_f32_16x16x32_bf16(
      /*neg_a=*/false, a, /*neg_b=*/false, b,
      /*c_mod=*/(short)0, c, /*reuse_a=*/false, /*reuse_b=*/false);
}

// A from bf16 buffer (LDS or global), row-major with leading dim ldA.
// B from bf16 row-major K x N (pre-transposed weights), leading dim ldB.
// unroll_count(4): enough in-flight loads for ILP without VGPR spills.
__device__ __forceinline__ v8f tile_mm_bf(const bf16* __restrict__ A, int ldA,
                                          const bf16* __restrict__ B, int ldB,
                                          int mb, int nb, int kIters,
                                          v8f acc, int lane) {
  const int koff = (lane >> 4) << 3;           // K offset per lane-half
  const bf16* arow = A + (mb + (lane & 15)) * ldA + koff;
  const bf16* brow = B + (long)koff * ldB + nb + (lane & 15);
#pragma clang loop unroll_count(4)
  for (int kk = 0; kk < kIters; ++kk) {
    ABf a;
    a.h.lo = *(const v8bf*)(arow);             // K = kb + koff .. +7   (VGPR 0-3)
    a.h.hi = *(const v8bf*)(arow + 16);        // K = kb + 16 + koff .. (VGPR 4-7)
    v16bf b;                                   // col N = lane&15, K = table order
#pragma unroll
    for (int s = 0; s < 8; ++s) { b[s] = brow[(long)s * ldB]; b[8 + s] = brow[(long)(16 + s) * ldB]; }
    acc = wmma_bf(a.v, b, acc);
    arow += 32;
    brow += (long)32 * ldB;
  }
  return acc;
}

// C-tile epilogue into a bf16 buffer. VGPR r -> row mb + r + 8*(lane>>4),
// col nb + (lane&15) per ISA C/D layout.
__device__ __forceinline__ void epilogue_bf(v8f acc, bf16* __restrict__ dst, int ldD,
                                            const float* __restrict__ bias,
                                            int mb, int nb, int lane, bool doTanh) {
  const int col = nb + (lane & 15);
  const int r0  = mb + ((lane >> 4) << 3);
  const float bv = bias[col];
#pragma unroll
  for (int r = 0; r < 8; ++r) {
    float v = acc[r] + bv;
    if (doTanh) v = tanhf(v);
    dst[(r0 + r) * ldD + col] = f2bf(v);
  }
}

// ---------------------------------------------------------------------------
// Weight conversion / transpose kernels (f32 -> bf16, stored K-major so WMMA
// B-tiles are contiguous per K-row)
// ---------------------------------------------------------------------------
__global__ void k_cvt_fW(const float* __restrict__ fW, bf16* __restrict__ fWt) {
  const long N = (long)(2 * H_) * H_;          // fWt[k][n] = fW[n][k], k in [0,2H)
  for (long i = (long)blockIdx.x * blockDim.x + threadIdx.x; i < N;
       i += (long)gridDim.x * blockDim.x) {
    long k = i >> 10, n = i & (H_ - 1);
    fWt[i] = f2bf(fW[n * (2 * H_) + k]);
  }
}

__global__ void k_cvt_nW1(const float* __restrict__ nW1, bf16* __restrict__ nW1f) {
  // comb = [out1|out1]  =>  fold: nW1f[k][n] = nW1[n][k] + nW1[n][H+k] (exact)
  const long N = (long)H_ * H_;
  for (long i = (long)blockIdx.x * blockDim.x + threadIdx.x; i < N;
       i += (long)gridDim.x * blockDim.x) {
    long k = i >> 10, n = i & (H_ - 1);
    nW1f[i] = f2bf(nW1[n * (2 * H_) + k] + nW1[n * (2 * H_) + H_ + k]);
  }
}

__global__ void k_cvt_nW2(const float* __restrict__ nW2, bf16* __restrict__ nW2t) {
  const long N = (long)H_ * H_;
  for (long i = (long)blockIdx.x * blockDim.x + threadIdx.x; i < N;
       i += (long)gridDim.x * blockDim.x) {
    long k = i >> 10, n = i & (H_ - 1);
    nW2t[i] = f2bf(nW2[n * H_ + k]);
  }
}

__global__ void k_cvt_logW(const float* __restrict__ logW, bf16* __restrict__ logWt) {
  const long N = (long)H_ * V_;                // logWt[k][v] = logW[v][k]
  for (long i = (long)blockIdx.x * blockDim.x + threadIdx.x; i < N;
       i += (long)gridDim.x * blockDim.x) {
    long k = i / V_, v = i - k * V_;
    logWt[i] = f2bf(logW[v * H_ + k]);
  }
}

// Pre-gather + convert all S*B embedding rows (4MB bf16, L2-resident).
// Removes the VALU-bound f32->bf16 path from the sequential recurrence.
__global__ void k_cvt_emb(const int* __restrict__ inputs,
                          const float* __restrict__ emb_table,
                          bf16* __restrict__ embs) {
  const long N = (long)S_ * B_ * H_;
  for (long i = (long)blockIdx.x * blockDim.x + threadIdx.x; i < N;
       i += (long)gridDim.x * blockDim.x) {
    long row = i >> 10, h = i & (H_ - 1);      // row = t*B + b
    int tok = inputs[row];
    embs[i] = f2bf(emb_table[(long)tok * H_ + h]);
  }
}

__global__ void k_copy_h1(const float* __restrict__ hin, float* __restrict__ hout) {
  // hidden[1] is never written by the reference loop -> pass through input
  int i = blockIdx.x * blockDim.x + threadIdx.x;
  if (i < B_ * H_) hout[B_ * H_ + i] = hin[B_ * H_ + i];
}

// ---------------------------------------------------------------------------
// Sequential recurrence: one 32-wave workgroup on one WGP.
// Activations ping-pong inside ONE shared array (offset toggle keeps the
// pointers provably in LDS addrspace -> ds_load_b128 in the hot loop).
// ---------------------------------------------------------------------------
constexpr int LDH = H_ + 8;                    // pad: break 256B LDS bank period

__global__ void __launch_bounds__(1024) k_recurrence(
    const bf16* __restrict__ embs, const float* __restrict__ hidden_in,
    const bf16* __restrict__ fWt,  const float* __restrict__ fb,
    const bf16* __restrict__ nW1f, const float* __restrict__ nb1,
    const bf16* __restrict__ nW2t, const float* __restrict__ nb2,
    bf16* __restrict__ out_all, float* __restrict__ hid_out) {
  __shared__ bf16 sBuf[2 * B_ * LDH];          // 2 x 66KB of the 320KB WGP LDS
  const int tid  = threadIdx.x;
  const int lane = tid & 31;
  const int wid  = tid >> 5;

  // h0 <- hidden[0] (f32 -> bf16) into buffer 0
  for (int i = tid; i < B_ * H_; i += 1024) {
    int b = i >> 10, h = i & (H_ - 1);
    sBuf[b * LDH + h] = f2bf(hidden_in[i]);
  }
  __syncthreads();

  int hOff = 0;                                // buffer holding h
  int tOff = B_ * LDH;                         // scratch / out1 / out3

#pragma clang loop unroll(disable)
  for (int t = 0; t < S_; ++t) {
    const bf16* bufH = sBuf + hOff;
    bf16*       bufT = sBuf + tOff;
    const bf16* embT = embs + (long)t * (B_ * H_);   // bf16 A rows, ld = H

    // GEMM1: out1 = [h | emb] @ fWt + fb  (no activation)  -> bufT
#pragma clang loop unroll(disable)
    for (int tix = wid; tix < 128; tix += 32) {
      const int mb = (tix & 1) << 4;
      const int nb = (tix >> 1) << 4;
      v8f acc = vzero();
      acc = tile_mm_bf(bufH, LDH, fWt, H_, mb, nb, H_ / 32, acc, lane);
      acc = tile_mm_bf(embT, H_, fWt + (size_t)H_ * H_, H_, mb, nb, H_ / 32, acc, lane);
      epilogue_bf(acc, bufT, LDH, fb, mb, nb, lane, /*tanh=*/false);
    }
    __syncthreads();

    // GEMM2: out2 = tanh(out1 @ nW1f + nb1)  (K folded 2H->H)  -> bufH slot
#pragma clang loop unroll(disable)
    for (int tix = wid; tix < 128; tix += 32) {
      const int mb = (tix & 1) << 4;
      const int nb = (tix >> 1) << 4;
      v8f acc = vzero();
      acc = tile_mm_bf(bufT, LDH, nW1f, H_, mb, nb, H_ / 32, acc, lane);
      epilogue_bf(acc, sBuf + hOff, LDH, nb1, mb, nb, lane, /*tanh=*/true);
    }
    __syncthreads();

    // GEMM3: out3 = tanh(out2 @ nW2t + nb2) -> bufT (= new h), out_all, hid_out
#pragma clang loop unroll(disable)
    for (int tix = wid; tix < 128; tix += 32) {
      const int mb = (tix & 1) << 4;
      const int nb = (tix >> 1) << 4;
      v8f acc = vzero();
      acc = tile_mm_bf(bufH, LDH, nW2t, H_, mb, nb, H_ / 32, acc, lane);
      const int col = nb + (lane & 15);
      const int r0  = mb + ((lane >> 4) << 3);
      const float bv = nb2[col];
#pragma unroll
      for (int r = 0; r < 8; ++r) {
        float v = tanhf(acc[r] + bv);
        bf16 bv16 = f2bf(v);
        bufT[(r0 + r) * LDH + col] = bv16;
        out_all[((long)t * B_ + r0 + r) * H_ + col] = bv16;
        if (t == S_ - 1) hid_out[(r0 + r) * H_ + col] = v;   // final hidden[0]
      }
    }
    __syncthreads();
    int tmp = hOff; hOff = tOff; tOff = tmp;   // new h now in bufT -> swap
  }
}

// ---------------------------------------------------------------------------
// Batched logits GEMM: (S*B=2048, H) @ (H, V) + logb -> f32.
// Block = 8 waves sharing one 64-column group; the 1024x64 bf16 B-slice
// (128KB) is staged into LDS by ONE Tensor-Data-Mover descriptor per block
// (tracked by TENSORcnt), with a 16B pad per 128B row -> 144B row stride
// (16B-aligned ds_load_b128, conflict-friendly). Each wave owns a 32x64
// register tile (8 f32 accumulators, no spills).
// ---------------------------------------------------------------------------
constexpr int BSTR = 72;                       // LDS B row stride in bf16 elems

__global__ void __launch_bounds__(256) k_logits(const bf16* __restrict__ out_all,
                                                const bf16* __restrict__ logWt,
                                                const float* __restrict__ logb,
                                                float* __restrict__ logits) {
  __shared__ bf16 sB[H_ * BSTR];               // 1024 rows x 144B = 147,456 B
  const int lane = threadIdx.x & 31;
  const int wid  = threadIdx.x >> 5;
  const int mb   = blockIdx.x / 500;           // 0..7   (8 x 256 rows)
  const int ng   = blockIdx.x - mb * 500;      // 0..499 (500 x 64 cols)
  const int n0   = ng * 64;
  const int m0   = mb * 256 + wid * 32;
  const int koff = (lane >> 4) << 3;
  const int ar   = lane & 15;

#if __has_builtin(__builtin_amdgcn_tensor_load_to_lds)
  if (wid == 0) {
    // Tensor DMA descriptor (ISA 8.3/8.4): 2D tile, data_size=2B,
    // tile 64(x) x 1024(y), tensor stride V, pad 4 dwords per 32 dwords.
    unsigned long long ga = (unsigned long long)(const void*)(logWt + n0);
    unsigned ldsA = (unsigned)(unsigned long long)(bf16_lds*)&sB[0];
    u32x4 g0 = {1u,                                            // count=1
                ldsA,                                          // lds_addr
                (unsigned)ga,                                  // global_addr lo
                (unsigned)((ga >> 32) & 0x1FFFFFFull) | 0x80000000u}; // hi|type=2
    i32x8 g1 = {(int)0x07110000u,            // data_size=1(2B) pad_en int=4 amt=3
                (int)(((unsigned)V_ & 0xFFFFu) << 16),         // tensor_dim0 lo16
                (int)(((unsigned)H_ & 0xFFFFu) << 16),         // tensor_dim1 lo16
                (int)(64u << 16),                              // tile_dim0 = 64
                H_,                                            // tile_dim1 = 1024
                V_,                                            // dim0_stride lo32
                0, 0};
    i32x4 z4 = {0, 0, 0, 0};
#if __clang_major__ >= 23
    i32x8 z8 = {0, 0, 0, 0, 0, 0, 0, 0};
    __builtin_amdgcn_tensor_load_to_lds(g0, g1, z4, z4, z8, 0);
#else
    __builtin_amdgcn_tensor_load_to_lds(g0, g1, z4, z4, 0);
#endif
    __builtin_amdgcn_s_wait_tensorcnt(0);
  }
#else
  // Fallback: cooperative synchronous staging of the B slice.
  for (int i = threadIdx.x; i < H_ * 8; i += 256) {
    int r = i >> 3, c = (i & 7) * 8;
    *(v8bf*)&sB[r * BSTR + c] = *(const v8bf*)(logWt + (long)r * V_ + n0 + c);
  }
#endif
  __syncthreads();

  v8f acc[2][4];
#pragma unroll
  for (int i = 0; i < 2; ++i)
#pragma unroll
    for (int j = 0; j < 4; ++j) acc[i][j] = vzero();

#pragma clang loop unroll_count(2)
  for (int kk = 0; kk < H_ / 32; ++kk) {
    const int kb = kk * 32;
    const bf16* brow = &sB[(kb + koff) * BSTR + ar];   // lane&15 = N col, K table order
    v16bf b[4];
#pragma unroll
    for (int j = 0; j < 4; ++j)
#pragma unroll
      for (int s = 0; s < 8; ++s) { b[j][s] = brow[s * BSTR + j * 16]; b[j][8 + s] = brow[(16 + s) * BSTR + j * 16]; }
#pragma unroll
    for (int i = 0; i < 2; ++i) {
      ABf a;
      const bf16* arow = out_all + (long)(m0 + i * 16 + ar) * H_ + kb;
      a.h.lo = *(const v8bf*)(arow + koff);
      a.h.hi = *(const v8bf*)(arow + 16 + koff);
      if (kk + 1 < H_ / 32) __builtin_prefetch(arow + 32, 0, 1);
#pragma unroll
      for (int j = 0; j < 4; ++j)
        acc[i][j] = wmma_bf(a.v, b[j], acc[i][j]);
    }
  }

  const int rb = (lane >> 4) << 3;
#pragma unroll
  for (int i = 0; i < 2; ++i)
#pragma unroll
    for (int j = 0; j < 4; ++j) {
      const int col = n0 + j * 16 + (lane & 15);
      const float bv = logb[col];
      float* orow = logits + (long)(m0 + i * 16 + rb) * V_ + col;
#pragma unroll
      for (int r = 0; r < 8; ++r) orow[(long)r * V_] = acc[i][j][r] + bv;
    }
}

// ---------------------------------------------------------------------------
// Launch
// ---------------------------------------------------------------------------
extern "C" void kernel_launch(void* const* d_in, const int* in_sizes, int n_in,
                              void* d_out, int out_size, void* d_ws, size_t ws_size,
                              hipStream_t stream) {
  (void)in_sizes; (void)n_in; (void)out_size; (void)ws_size;

  const int*   inputs    = (const int*)d_in[0];
  const float* hidden    = (const float*)d_in[1];
  const float* emb_table = (const float*)d_in[2];
  const float* fW   = (const float*)d_in[3];
  const float* fb   = (const float*)d_in[4];
  const float* nW1  = (const float*)d_in[5];
  const float* nb1  = (const float*)d_in[6];
  const float* nW2  = (const float*)d_in[7];
  const float* nb2  = (const float*)d_in[8];
  const float* logW = (const float*)d_in[9];
  const float* logb = (const float*)d_in[10];

  float* logits  = (float*)d_out;
  float* hid_out = logits + (size_t)S_ * B_ * V_;

  // Workspace layout (bf16 elements, all offsets 32B-aligned): ~78.5 MB total
  bf16* wsb     = (bf16*)d_ws;
  bf16* fWt     = wsb;                               // (2H x H)
  bf16* nW1f    = fWt  + (size_t)(2 * H_) * H_;      // (H x H), folded
  bf16* nW2t    = nW1f + (size_t)H_ * H_;            // (H x H)
  bf16* logWt   = nW2t + (size_t)H_ * H_;            // (H x V)
  bf16* out_all = logWt + (size_t)H_ * V_;           // (S*B x H)
  bf16* embs    = out_all + (size_t)S_ * B_ * H_;    // (S*B x H)

  k_cvt_fW  <<<1024, 256, 0, stream>>>(fW,  fWt);
  k_cvt_nW1 <<<1024, 256, 0, stream>>>(nW1, nW1f);
  k_cvt_nW2 <<<1024, 256, 0, stream>>>(nW2, nW2t);
  k_cvt_logW<<<4096, 256, 0, stream>>>(logW, logWt);
  k_cvt_emb <<<1024, 256, 0, stream>>>(inputs, emb_table, embs);
  k_copy_h1 <<<64, 512, 0, stream>>>(hidden, hid_out);

  k_recurrence<<<1, 1024, 0, stream>>>(embs, hidden,
                                       fWt, fb, nW1f, nb1, nW2t, nb2,
                                       out_all, hid_out);

  k_logits<<<4000, 256, 0, stream>>>(out_all, logWt, logb, logits);
}
